// TransformerBlock_82403242541511
// MI455X (gfx1250) — matrix operand order repaired and checked by reading, not compiled, not run on hardware
//
#include <hip/hip_runtime.h>

// ---------------------------------------------------------------------------
// Point-Transformer block for MI455X (gfx1250, wave32, WMMA 16x16x32 f16)
// ---------------------------------------------------------------------------
#define BATCH   4
#define NPTS    4096
#define KNN     16
#define DPTS    64
#define DM      128
#define PPB     32          // points per block in point_kernel

typedef _Float16 half_t;
typedef __attribute__((ext_vector_type(16))) _Float16 v16h;
typedef __attribute__((ext_vector_type(8)))  float    v8f;

// Load a 16x32 f16 A-fragment from row-major LDS (lda halves per row).
// Layout: lanes 0-15: M=lane, K = k0 + {0..15};
//         lanes 16-31: M=lane-16, K = k0 + {16..31}.
__device__ __forceinline__ v16h load_a16(const half_t* a, int lda, int k0, int lane) {
    const int m  = lane & 15;
    const int kb = (lane & 16) ? 16 : 0;
    const half_t* p = a + m * lda + k0 + kb;
    v16h r;
#pragma unroll
    for (int i = 0; i < 16; ++i) { r[i] = p[i]; }
    return r;
}

// ---------------------------------------------------------------------------
// Kernel 1: convert weights f32->f16 into pre-swizzled WMMA B-fragment order.
// For matrix W (KxN row-major, N==128): fragment chunk c = nt*ksteps + ks holds
// per-lane v16h: lane l -> col n = nt*16+(l&15), K = ks*32 + ((l&16)?16:0) + i.
// ---------------------------------------------------------------------------
__global__ void __launch_bounds__(256) prep_weights(
    const float* __restrict__ wfc1, const float* __restrict__ wq,
    const float* __restrict__ wk,   const float* __restrict__ wv,
    const float* __restrict__ wd2,  const float* __restrict__ wg1,
    const float* __restrict__ wg2,
    half_t* __restrict__ ffc1, half_t* __restrict__ fq,
    half_t* __restrict__ fk,   half_t* __restrict__ fv,
    half_t* __restrict__ fd2,  half_t* __restrict__ fg1,
    half_t* __restrict__ fg2)
{
    const int blk = blockIdx.x;              // 0..207
    const float* src; half_t* dst; int ksteps, chunk;
    if (blk < 16) { src = wfc1; dst = ffc1; ksteps = 2; chunk = blk; }
    else {
        const int r = blk - 16; const int m = r >> 5; chunk = r & 31; ksteps = 4;
        switch (m) {
            case 0:  src = wq;  dst = fq;  break;
            case 1:  src = wk;  dst = fk;  break;
            case 2:  src = wv;  dst = fv;  break;
            case 3:  src = wd2; dst = fd2; break;
            case 4:  src = wg1; dst = fg1; break;
            default: src = wg2; dst = fg2; break;
        }
    }
    const int nt = chunk / ksteps, ks = chunk % ksteps;
    for (int t = threadIdx.x; t < 512; t += 256) {
        const int l = t >> 4, i = t & 15;
        const int n = nt * 16 + (l & 15);
        const int k = ks * 32 + ((l & 16) ? 16 : 0) + i;
        dst[chunk * 512 + t] = (half_t)src[k * DM + n];
    }
}

// ---------------------------------------------------------------------------
// Kernel 2: kNN (K=16 smallest squared distances, self included, ascending).
// One thread per query point; candidates staged through LDS in 256-pt tiles.
// ---------------------------------------------------------------------------
__global__ void __launch_bounds__(256) knn_kernel(
    const float* __restrict__ xyz, int* __restrict__ knn_idx)
{
    const int b     = blockIdx.x >> 4;            // NPTS/256 == 16 blocks per batch
    const int qi    = ((blockIdx.x & 15) << 8) + threadIdx.x;
    const float* xb = xyz + (size_t)b * NPTS * 3;
    const float qx = xb[qi * 3 + 0], qy = xb[qi * 3 + 1], qz = xb[qi * 3 + 2];

    float bd[KNN]; int bi[KNN];
#pragma unroll
    for (int j = 0; j < KNN; ++j) { bd[j] = 3.4e38f; bi[j] = 0; }

    __shared__ float sx[256], sy[256], sz[256];
    for (int t = 0; t < NPTS; t += 256) {
        __syncthreads();
        const int m = t + threadIdx.x;
        sx[threadIdx.x] = xb[m * 3 + 0];
        sy[threadIdx.x] = xb[m * 3 + 1];
        sz[threadIdx.x] = xb[m * 3 + 2];
        __syncthreads();
        for (int j = 0; j < 256; ++j) {
            const float dx = qx - sx[j], dy = qy - sy[j], dz = qz - sz[j];
            const float d = dx * dx + dy * dy + dz * dz;
            if (d < bd[KNN - 1]) {
                bd[KNN - 1] = d; bi[KNN - 1] = t + j;
#pragma unroll
                for (int r = KNN - 1; r > 0; --r) {
                    if (bd[r] < bd[r - 1]) {
                        const float td = bd[r]; bd[r] = bd[r - 1]; bd[r - 1] = td;
                        const int   ti = bi[r]; bi[r] = bi[r - 1]; bi[r - 1] = ti;
                    }
                }
            }
        }
    }
    int* out = knn_idx + ((size_t)b * NPTS + qi) * KNN;
#pragma unroll
    for (int j = 0; j < KNN; ++j) out[j] = bi[j];
}

// ---------------------------------------------------------------------------
// Kernel 3: x = feat@w_fc1 + b_fc1 (f16 in LDS), then q/k/v = x@{wq,wk,wv}
// 64 rows per block, 8 waves, WMMA f16 -> f32 results to workspace.
// ---------------------------------------------------------------------------
__global__ void __launch_bounds__(256) proj_kernel(
    const float* __restrict__ feat, const float* __restrict__ b_fc1,
    const half_t* __restrict__ ffc1, const half_t* __restrict__ fqw,
    const half_t* __restrict__ fkw,  const half_t* __restrict__ fvw,
    float* __restrict__ qo, float* __restrict__ ko, float* __restrict__ vo)
{
    __shared__ half_t sA[64 * DPTS];   // features tile, f16
    __shared__ half_t sX[64 * DM];     // x tile, f16
    __shared__ float  sB1[DM];
    const int tid = threadIdx.x;
    const int lane = tid & 31, wv = tid >> 5;
    const int rowbase = blockIdx.x * 64;

    for (int t = tid; t < 64 * DPTS; t += 256)
        sA[t] = (half_t)feat[(size_t)(rowbase + (t >> 6)) * DPTS + (t & 63)];
    if (tid < DM) sB1[tid] = b_fc1[tid];
    __syncthreads();

    // x = feat @ w_fc1 + b_fc1
#pragma unroll
    for (int t = 0; t < 4; ++t) {
        const int tile = wv * 4 + t;
        const int mt = tile >> 3, nt = tile & 7;
        v8f acc = {};
#pragma unroll
        for (int ks = 0; ks < 2; ++ks) {
            v16h a  = load_a16(sA + mt * 16 * DPTS, DPTS, ks * 32, lane);
            v16h bf = *(const v16h*)(ffc1 + (size_t)((nt * 2 + ks) * 32 + lane) * 16);
            acc = __builtin_amdgcn_wmma_f32_16x16x32_f16(false, a, false, bf,
                                                         (short)0, acc, false, false);
        }
        const int col = nt * 16 + (lane & 15);
        const int mof = (lane & 16) ? 8 : 0;
        const float bias = sB1[col];
#pragma unroll
        for (int r = 0; r < 8; ++r)
            sX[(mt * 16 + mof + r) * DM + col] = (half_t)(acc[r] + bias);
    }
    __syncthreads();

    const half_t* wf[3] = { fqw, fkw, fvw };
    float*        op[3] = { qo, ko, vo };
#pragma unroll
    for (int mm = 0; mm < 3; ++mm) {
#pragma unroll
        for (int t = 0; t < 4; ++t) {
            const int tile = wv * 4 + t;
            const int mt = tile >> 3, nt = tile & 7;
            v8f acc = {};
#pragma unroll
            for (int ks = 0; ks < 4; ++ks) {
                v16h a  = load_a16(sX + mt * 16 * DM, DM, ks * 32, lane);
                v16h bf = *(const v16h*)(wf[mm] + (size_t)((nt * 4 + ks) * 32 + lane) * 16);
                acc = __builtin_amdgcn_wmma_f32_16x16x32_f16(false, a, false, bf,
                                                             (short)0, acc, false, false);
            }
            const int col = nt * 16 + (lane & 15);
            const int mof = (lane & 16) ? 8 : 0;
            float* o = op[mm] + (size_t)(rowbase + mt * 16 + mof) * DM + col;
#pragma unroll
            for (int r = 0; r < 8; ++r) o[(size_t)r * DM] = acc[r];
        }
    }
}

// ---------------------------------------------------------------------------
// Kernel 4: fused per-point attention. 4 waves/point, each wave owns 2 N-tiles.
// B fragments for w_d2/w_g1/w_g2 are hoisted into registers once per block
// (24 x v16h = 192 VGPRs) since every point reuses them -> the GEMM inner
// loops become pure ds_load_b128 (A) + v_wmma. attn goes out with NT stores
// so the 134 MB single-use stream doesn't evict the L2-resident k/v buffers.
// ---------------------------------------------------------------------------
__global__ void __launch_bounds__(128) point_kernel(
    const float* __restrict__ xyz,  const float* __restrict__ feat,
    const int*   __restrict__ knn,  const float* __restrict__ qb,
    const float* __restrict__ kbuf, const float* __restrict__ vbuf,
    const half_t* __restrict__ fd2, const half_t* __restrict__ fg1,
    const half_t* __restrict__ fg2,
    const float* __restrict__ w_d1, const float* __restrict__ b_d1,
    const float* __restrict__ b_d2, const float* __restrict__ b_g1,
    const float* __restrict__ b_g2,
    const float* __restrict__ w_fc2, const float* __restrict__ b_fc2,
    float* __restrict__ out_res, float* __restrict__ out_attn)
{
    __shared__ float  s_wd1[3 * DM];
    __shared__ float  s_bd1[DM], s_bd2[DM], s_bg1[DM], s_bg2[DM];
    __shared__ float  s_q[DM];
    __shared__ float  s_rel[KNN * 3];
    __shared__ int    s_idx[KNN];
    __shared__ half_t s_h[KNN * DM];
    __shared__ half_t s_t[KNN * DM];
    __shared__ half_t s_a1[KNN * DM];
    __shared__ float  s_pe[KNN * DM];
    __shared__ float  s_res[DM];

    const int tid  = threadIdx.x;
    const int lane = tid & 31, wv = tid >> 5;
    const float SCALE = 0.088388347648318447f;   // 1/sqrt(128)

    // ---- hoist loop-invariant B fragments into registers ----
    v16h Bd2[2][4], Bg1[2][4], Bg2[2][4];
#pragma unroll
    for (int tt = 0; tt < 2; ++tt) {
        const int nt = wv * 2 + tt;
#pragma unroll
        for (int ks = 0; ks < 4; ++ks) {
            const size_t fi = (size_t)((nt * 4 + ks) * 32 + lane) * 16;
            Bd2[tt][ks] = *(const v16h*)(fd2 + fi);
            Bg1[tt][ks] = *(const v16h*)(fg1 + fi);
            Bg2[tt][ks] = *(const v16h*)(fg2 + fi);
        }
    }

    for (int t = tid; t < 3 * DM; t += 128) s_wd1[t] = w_d1[t];
    if (tid < DM) {
        s_bd1[tid] = b_d1[tid]; s_bd2[tid] = b_d2[tid];
        s_bg1[tid] = b_g1[tid];
        s_bg2[tid] = b_g2[tid] * SCALE;          // pre-scaled for softmax
    }

    for (int p = 0; p < PPB; ++p) {
        const int pt = blockIdx.x * PPB + p;
        const int b  = pt >> 12;
        __syncthreads();
        if (tid < DM) s_q[tid] = qb[(size_t)pt * DM + tid];
        if (tid < KNN) {
            const int idx = knn[(size_t)pt * KNN + tid];
            s_idx[tid] = idx;
            const float* xp = xyz + (size_t)pt * 3;
            const float* xn = xyz + ((size_t)b * NPTS + idx) * 3;
            s_rel[tid * 3 + 0] = xp[0] - xn[0];
            s_rel[tid * 3 + 1] = xp[1] - xn[1];
            s_rel[tid * 3 + 2] = xp[2] - xn[2];
        }
        __syncthreads();

        // h = relu(rel @ w_d1 + b_d1)   (K-dim = 3, VALU)
        for (int t = tid; t < KNN * DM; t += 128) {
            const int m = t >> 7, c = t & (DM - 1);
            float v = s_bd1[c]
                    + s_rel[m * 3 + 0] * s_wd1[0 * DM + c]
                    + s_rel[m * 3 + 1] * s_wd1[1 * DM + c]
                    + s_rel[m * 3 + 2] * s_wd1[2 * DM + c];
            s_h[t] = (half_t)(v > 0.f ? v : 0.f);
        }
        __syncthreads();

        // pos_enc = h @ w_d2 + b_d2 ;  t = q - k_g + pos_enc
#pragma unroll
        for (int tt = 0; tt < 2; ++tt) {
            const int nt = wv * 2 + tt;
            v8f acc = {};
#pragma unroll
            for (int ks = 0; ks < 4; ++ks) {
                v16h a = load_a16(s_h, DM, ks * 32, lane);
                acc = __builtin_amdgcn_wmma_f32_16x16x32_f16(false, a, false, Bd2[tt][ks],
                                                             (short)0, acc, false, false);
            }
            const int col = nt * 16 + (lane & 15);
            const int mof = (lane & 16) ? 8 : 0;
#pragma unroll
            for (int r = 0; r < 8; ++r) {
                const int m = mof + r;
                const float pe = acc[r] + s_bd2[col];
                s_pe[m * DM + col] = pe;
                const float kv = kbuf[((size_t)b * NPTS + s_idx[m]) * DM + col];
                s_t[m * DM + col] = (half_t)(s_q[col] - kv + pe);
            }
        }
        __syncthreads();

        // a1 = relu(t @ w_g1 + b_g1)
#pragma unroll
        for (int tt = 0; tt < 2; ++tt) {
            const int nt = wv * 2 + tt;
            v8f acc = {};
#pragma unroll
            for (int ks = 0; ks < 4; ++ks) {
                v16h a = load_a16(s_t, DM, ks * 32, lane);
                acc = __builtin_amdgcn_wmma_f32_16x16x32_f16(false, a, false, Bg1[tt][ks],
                                                             (short)0, acc, false, false);
            }
            const int col = nt * 16 + (lane & 15);
            const int mof = (lane & 16) ? 8 : 0;
#pragma unroll
            for (int r = 0; r < 8; ++r) {
                const float v = acc[r] + s_bg1[col];
                s_a1[(mof + r) * DM + col] = (half_t)(v > 0.f ? v : 0.f);
            }
        }
        __syncthreads();

        // logits = (a1 @ w_g2 + b_g2)/sqrt(D) ; softmax over 16 neighbor rows
#pragma unroll
        for (int tt = 0; tt < 2; ++tt) {
            const int nt = wv * 2 + tt;
            v8f acc = {};
#pragma unroll
            for (int ks = 0; ks < 4; ++ks) {
                v16h a = load_a16(s_a1, DM, ks * 32, lane);
                acc = __builtin_amdgcn_wmma_f32_16x16x32_f16(false, a, false, Bg2[tt][ks],
                                                             (short)0, acc, false, false);
            }
            const int col = nt * 16 + (lane & 15);
            const int mof = (lane & 16) ? 8 : 0;
            float lg[8];
            float mx = -3.4e38f;
#pragma unroll
            for (int r = 0; r < 8; ++r) {
                lg[r] = fmaf(acc[r], SCALE, s_bg2[col]);
                mx = fmaxf(mx, lg[r]);
            }
            mx = fmaxf(mx, __shfl_xor(mx, 16, 32));   // join lanes l <-> l+16
            float sum = 0.f;
#pragma unroll
            for (int r = 0; r < 8; ++r) { lg[r] = __expf(lg[r] - mx); sum += lg[r]; }
            sum += __shfl_xor(sum, 16, 32);
            const float inv = 1.f / sum;
            float wsum = 0.f;
#pragma unroll
            for (int r = 0; r < 8; ++r) {
                const int m = mof + r;
                const float a = lg[r] * inv;
                __builtin_nontemporal_store(a, out_attn + ((size_t)pt * KNN + m) * DM + col);
                const float vv = vbuf[((size_t)b * NPTS + s_idx[m]) * DM + col]
                               + s_pe[m * DM + col];
                wsum += a * vv;
            }
            wsum += __shfl_xor(wsum, 16, 32);
            if (lane < 16) s_res[col] = wsum;
        }
        __syncthreads();

        // res = s_res @ w_fc2 + b_fc2 + features
        if (tid < DPTS) {
            float acc = b_fc2[tid];
#pragma unroll 4
            for (int k2 = 0; k2 < DM; ++k2) acc += s_res[k2] * w_fc2[k2 * DPTS + tid];
            out_res[(size_t)pt * DPTS + tid] = acc + feat[(size_t)pt * DPTS + tid];
        }
    }
}

// ---------------------------------------------------------------------------
// Host launcher
// ---------------------------------------------------------------------------
extern "C" void kernel_launch(void* const* d_in, const int* in_sizes, int n_in,
                              void* d_out, int out_size, void* d_ws, size_t ws_size,
                              hipStream_t stream) {
    (void)in_sizes; (void)n_in; (void)out_size; (void)ws_size;

    const float* xyz    = (const float*)d_in[0];
    const float* feat   = (const float*)d_in[1];
    const float* w_fc1  = (const float*)d_in[2];
    const float* b_fc1  = (const float*)d_in[3];
    const float* w_fc2  = (const float*)d_in[4];
    const float* b_fc2  = (const float*)d_in[5];
    const float* w_d1   = (const float*)d_in[6];
    const float* b_d1   = (const float*)d_in[7];
    const float* w_d2   = (const float*)d_in[8];
    const float* b_d2   = (const float*)d_in[9];
    const float* w_g1   = (const float*)d_in[10];
    const float* b_g1   = (const float*)d_in[11];
    const float* w_g2   = (const float*)d_in[12];
    const float* b_g2   = (const float*)d_in[13];
    const float* w_q    = (const float*)d_in[14];
    const float* w_k    = (const float*)d_in[15];
    const float* w_v    = (const float*)d_in[16];

    char* ws = (char*)d_ws;
    float*  qb   = (float*)(ws + 0);          // B*N*128 f32 = 8 MiB
    float*  kbf  = (float*)(ws + 8388608);    // 8 MiB
    float*  vbf  = (float*)(ws + 16777216);   // 8 MiB
    int*    ib   = (int*)  (ws + 25165824);   // B*N*16 i32 = 1 MiB
    half_t* ffc1 = (half_t*)(ws + 26214400);  // 16 KiB
    half_t* fq   = (half_t*)(ws + 26230784);  // 32 KiB
    half_t* fk   = (half_t*)(ws + 26263552);
    half_t* fv   = (half_t*)(ws + 26296320);
    half_t* fd2  = (half_t*)(ws + 26329088);
    half_t* fg1  = (half_t*)(ws + 26361856);
    half_t* fg2  = (half_t*)(ws + 26394624);  // end @ ~25.2 MiB

    float* out_res  = (float*)d_out;                         // (B,N,64)
    float* out_attn = out_res + (size_t)BATCH * NPTS * DPTS; // (B,N,16,128)

    prep_weights<<<208, 256, 0, stream>>>(w_fc1, w_q, w_k, w_v, w_d2, w_g1, w_g2,
                                          ffc1, fq, fk, fv, fd2, fg1, fg2);
    knn_kernel<<<BATCH * (NPTS / 256), 256, 0, stream>>>(xyz, ib);
    proj_kernel<<<(BATCH * NPTS) / 64, 256, 0, stream>>>(feat, b_fc1, ffc1, fq, fk, fv,
                                                         qb, kbf, vbf);
    point_kernel<<<(BATCH * NPTS) / PPB, 128, 0, stream>>>(
        xyz, feat, ib, qb, kbf, vbf, fd2, fg1, fg2,
        w_d1, b_d1, b_d2, b_g1, b_g2, w_fc2, b_fc2, out_res, out_attn);
}
